// SpatialSelfAttention_47184510714029
// MI455X (gfx1250) — matrix operand order repaired and checked by reading, not compiled, not run on hardware
//
#include <hip/hip_runtime.h>
#include <hip/hip_bf16.h>

// ---------------------------------------------------------------------------
// Spatial self-attention (b=16, c=512, n=2048) for MI455X / gfx1250.
// All six GEMMs through v_wmma_f32_16x16x32_bf16, double-buffered LDS with
// GLOBAL_LOAD_ASYNC_TO_LDS (ASYNCcnt) staging when the builtin exists.
// ---------------------------------------------------------------------------

typedef __attribute__((ext_vector_type(16))) __bf16 v16bf;
typedef __attribute__((ext_vector_type(8)))  __bf16 v8bf;
typedef __attribute__((ext_vector_type(8)))  float  v8f;
typedef __attribute__((ext_vector_type(4)))  int    v4i;

#define BATCH 16
#define CDIM  512
#define NDIM  2048
#define GROUPS 32
#define CPG   16

#if defined(__has_builtin)
#if __has_builtin(__builtin_amdgcn_global_load_async_to_lds_b128)
#define USE_ASYNC_COPY 1
#endif
#endif

#define GAS __attribute__((address_space(1)))
#define LAS __attribute__((address_space(3)))

__device__ __forceinline__ void async_wait0() {
#if defined(__has_builtin) && __has_builtin(__builtin_amdgcn_s_wait_asynccnt)
    __builtin_amdgcn_s_wait_asynccnt(0);
#else
    asm volatile("s_wait_asynccnt 0x0" ::: "memory");
#endif
}

#ifdef USE_ASYNC_COPY
__device__ __forceinline__ void async_cp16(const __bf16* g, __bf16* l) {
    __builtin_amdgcn_global_load_async_to_lds_b128(
        (GAS v4i*)(void*)g, (LAS v4i*)(void*)l, 0, 0);
}
#endif

// ------------------------------ K1: weight convert -------------------------
__global__ void wconv_kernel(const float* __restrict__ wq, const float* __restrict__ wk,
                             const float* __restrict__ wv, const float* __restrict__ wo,
                             __bf16* __restrict__ out, int per) {
    int i = blockIdx.x * blockDim.x + threadIdx.x;
    if (i >= 4 * per) return;
    int which = i / per;
    int idx = i - which * per;
    const float* src = (which == 0) ? wq : (which == 1) ? wk : (which == 2) ? wv : wo;
    out[i] = (__bf16)src[idx];
}

// ------------------------- K2: GroupNorm + transpose -----------------------
__global__ __launch_bounds__(256)
void groupnorm_kernel(const float* __restrict__ x, const float* __restrict__ gamma,
                      const float* __restrict__ beta, __bf16* __restrict__ h_t) {
    const int g = blockIdx.x;
    const int b = blockIdx.y;
    const int t = threadIdx.x;
    const int cbase = g * CPG;
    const float* xb = x + ((size_t)b * CDIM + cbase) * NDIM;

    __shared__ float rs[256];
    __shared__ float rss[256];
    __shared__ float tile[64 * 17];

    const int total = CPG * NDIM;
    float s = 0.f, ss = 0.f;
    for (int i = t; i < total; i += 256) {
        float v = xb[i];
        s += v; ss += v * v;
    }
    rs[t] = s; rss[t] = ss;
    __syncthreads();
    for (int off = 128; off > 0; off >>= 1) {
        if (t < off) { rs[t] += rs[t + off]; rss[t] += rss[t + off]; }
        __syncthreads();
    }
    const float mean = rs[0] * (1.f / (float)total);
    const float var  = rss[0] * (1.f / (float)total) - mean * mean;
    const float rstd = rsqrtf(var + 1e-6f);
    __syncthreads();

    const int cl = t >> 6;
    const int nl = t & 63;
    const int c2 = t & 15;
    const int n2 = t >> 4;
    for (int nc = 0; nc < NDIM; nc += 64) {
#pragma unroll
        for (int i = 0; i < 4; i++) {
            int c = cl + i * 4;
            float v = xb[(size_t)c * NDIM + nc + nl];
            tile[nl * 17 + c] = (v - mean) * rstd * gamma[cbase + c] + beta[cbase + c];
        }
        __syncthreads();
#pragma unroll
        for (int i = 0; i < 4; i++) {
            int n = n2 + i * 16;
            h_t[((size_t)b * NDIM + nc + n) * CDIM + cbase + c2] = (__bf16)tile[n * 17 + c2];
        }
        __syncthreads();
    }
}

// ------------------------------ generic WMMA GEMM --------------------------
// C[M][N] = alpha*(A[M][K] @ B[K][N] + bias[M]) (+resid, fp32). Bt is [N][K].
// BM=128, BN=256, BK=32; 8 waves of 64x64 tiles; double-buffered LDS,
// one barrier per K-step, async global->LDS staging when available.
#define BM 128
#define BN 256
#define BK 32
#define LDT 40   // padded LDS row stride in bf16 elems (80B)

__global__ __launch_bounds__(256)
void gemm_bf16_kernel(const __bf16* __restrict__ A, size_t sA,
                      const __bf16* __restrict__ Bt, size_t sB,
                      int M, int N, int K,
                      const float* __restrict__ bias, float alpha,
                      __bf16* __restrict__ Cb, float* __restrict__ Cf,
                      const float* __restrict__ resid,
                      size_t sC, size_t sR, int transC) {
    __shared__ __bf16 As[2][BM * LDT];   // 2 x 10.0 KB
    __shared__ __bf16 Bs[2][BN * LDT];   // 2 x 20.0 KB

    const int t    = threadIdx.x;
    const int bz   = blockIdx.z;
    const int m0   = blockIdx.y * BM;
    const int n0   = blockIdx.x * BN;
    const __bf16* Ab = A  + (size_t)bz * sA;
    const __bf16* Bb = Bt + (size_t)bz * sB;

    const int wave = t >> 5;
    const int lane = t & 31;
    const int wm   = (wave >> 2) * 64;   // 0 / 64
    const int wn   = (wave & 3)  * 64;   // 0..192
    const int ln   = lane & 15;
    const int lh   = lane >> 4;

    v8f acc[4][4];
#pragma unroll
    for (int i = 0; i < 4; i++)
#pragma unroll
        for (int j = 0; j < 4; j++)
#pragma unroll
            for (int r = 0; r < 8; r++) acc[i][j][r] = 0.f;

    // staging map: thread t -> A row sr (k half sc), B rows sr and sr+128
    const int sr = t >> 1;            // 0..127
    const int sc = (t & 1) * 16;      // 0 / 16
    const int steps = K / BK;

#ifdef USE_ASYNC_COPY
    {   // prologue: tile 0 -> buffer 0
        const size_t aoff = (size_t)(m0 + sr) * K + sc;
        const size_t b0   = (size_t)(n0 + sr) * K + sc;
        const size_t b1   = (size_t)(n0 + sr + 128) * K + sc;
        async_cp16(&Ab[aoff],     &As[0][sr * LDT + sc]);
        async_cp16(&Ab[aoff + 8], &As[0][sr * LDT + sc + 8]);
        async_cp16(&Bb[b0],       &Bs[0][sr * LDT + sc]);
        async_cp16(&Bb[b0 + 8],   &Bs[0][sr * LDT + sc + 8]);
        async_cp16(&Bb[b1],       &Bs[0][(sr + 128) * LDT + sc]);
        async_cp16(&Bb[b1 + 8],   &Bs[0][(sr + 128) * LDT + sc + 8]);
    }
#else
    {   // prologue: load tile 0, store to buffer 0
        const size_t aoff = (size_t)(m0 + sr) * K + sc;
        const size_t b0   = (size_t)(n0 + sr) * K + sc;
        const size_t b1   = (size_t)(n0 + sr + 128) * K + sc;
        v8bf a0 = *(const v8bf*)&Ab[aoff];
        v8bf a1 = *(const v8bf*)&Ab[aoff + 8];
        v8bf q0 = *(const v8bf*)&Bb[b0];
        v8bf q1 = *(const v8bf*)&Bb[b0 + 8];
        v8bf q2 = *(const v8bf*)&Bb[b1];
        v8bf q3 = *(const v8bf*)&Bb[b1 + 8];
        *(v8bf*)&As[0][sr * LDT + sc]           = a0;
        *(v8bf*)&As[0][sr * LDT + sc + 8]       = a1;
        *(v8bf*)&Bs[0][sr * LDT + sc]           = q0;
        *(v8bf*)&Bs[0][sr * LDT + sc + 8]       = q1;
        *(v8bf*)&Bs[0][(sr + 128) * LDT + sc]   = q2;
        *(v8bf*)&Bs[0][(sr + 128) * LDT + sc + 8] = q3;
    }
#endif

    for (int kt = 0; kt < steps; kt++) {
        const int cur = kt & 1;
        const int nxt = cur ^ 1;
        const int kn  = (kt + 1) * BK;
        const bool more = (kt + 1) < steps;

#ifdef USE_ASYNC_COPY
        async_wait0();                 // my tile-kt copies landed in LDS
        __syncthreads();               // everyone's landed; prev reads of nxt done
        if (more) {                    // overlap next tile copy with compute
            const size_t aoff = (size_t)(m0 + sr) * K + kn + sc;
            const size_t b0   = (size_t)(n0 + sr) * K + kn + sc;
            const size_t b1   = (size_t)(n0 + sr + 128) * K + kn + sc;
            async_cp16(&Ab[aoff],     &As[nxt][sr * LDT + sc]);
            async_cp16(&Ab[aoff + 8], &As[nxt][sr * LDT + sc + 8]);
            async_cp16(&Bb[b0],       &Bs[nxt][sr * LDT + sc]);
            async_cp16(&Bb[b0 + 8],   &Bs[nxt][sr * LDT + sc + 8]);
            async_cp16(&Bb[b1],       &Bs[nxt][(sr + 128) * LDT + sc]);
            async_cp16(&Bb[b1 + 8],   &Bs[nxt][(sr + 128) * LDT + sc + 8]);
        }
#else
        __syncthreads();
        v8bf a0, a1, q0, q1, q2, q3;
        if (more) {                    // prefetch next tile to regs during compute
            const size_t aoff = (size_t)(m0 + sr) * K + kn + sc;
            const size_t b0   = (size_t)(n0 + sr) * K + kn + sc;
            const size_t b1   = (size_t)(n0 + sr + 128) * K + kn + sc;
            a0 = *(const v8bf*)&Ab[aoff];
            a1 = *(const v8bf*)&Ab[aoff + 8];
            q0 = *(const v8bf*)&Bb[b0];
            q1 = *(const v8bf*)&Bb[b0 + 8];
            q2 = *(const v8bf*)&Bb[b1];
            q3 = *(const v8bf*)&Bb[b1 + 8];
        }
#endif

        // fragments from buffer `cur`
        v16bf af[4];
#pragma unroll
        for (int i = 0; i < 4; i++) {
            int row = wm + i * 16 + ln;
            v8bf lo = *(const v8bf*)&As[cur][row * LDT + lh * 8];
            v8bf hi = *(const v8bf*)&As[cur][row * LDT + 16 + lh * 8];
            af[i] = __builtin_shufflevector(lo, hi, 0, 1, 2, 3, 4, 5, 6, 7,
                                            8, 9, 10, 11, 12, 13, 14, 15);
        }
        v16bf bfv[4];
#pragma unroll
        for (int j = 0; j < 4; j++) {
            int row = wn + j * 16 + ln;
            v8bf lo = *(const v8bf*)&Bs[cur][row * LDT + lh * 8];
            v8bf hi = *(const v8bf*)&Bs[cur][row * LDT + 16 + lh * 8];
            bfv[j] = __builtin_shufflevector(lo, hi, 0, 1, 2, 3, 4, 5, 6, 7,
                                             8, 9, 10, 11, 12, 13, 14, 15);
        }
#pragma unroll
        for (int i = 0; i < 4; i++)
#pragma unroll
            for (int j = 0; j < 4; j++)
                acc[i][j] = __builtin_amdgcn_wmma_f32_16x16x32_bf16(
                    false, af[i], false, bfv[j], (short)0, acc[i][j], false, false);

#ifndef USE_ASYNC_COPY
        if (more) {
            *(v8bf*)&As[nxt][sr * LDT + sc]             = a0;
            *(v8bf*)&As[nxt][sr * LDT + sc + 8]         = a1;
            *(v8bf*)&Bs[nxt][sr * LDT + sc]             = q0;
            *(v8bf*)&Bs[nxt][sr * LDT + sc + 8]         = q1;
            *(v8bf*)&Bs[nxt][(sr + 128) * LDT + sc]     = q2;
            *(v8bf*)&Bs[nxt][(sr + 128) * LDT + sc + 8] = q3;
        }
#endif
    }

    // ---- epilogue ----
#pragma unroll
    for (int i = 0; i < 4; i++) {
#pragma unroll
        for (int j = 0; j < 4; j++) {
            const int mbase = m0 + wm + i * 16 + lh * 8;   // 8 m-contiguous per lane
            const int nn    = n0 + wn + j * 16 + ln;
            if (Cf) {
#pragma unroll
                for (int r = 0; r < 8; r++) {
                    int m = mbase + r;
                    float v = alpha * (acc[i][j][r] + (bias ? bias[m] : 0.f));
                    if (resid) v += resid[(size_t)bz * sR + (size_t)m * N + nn];
                    Cf[(size_t)bz * sC + (size_t)m * N + nn] = v;
                }
            } else if (transC) {
                v8bf pack;
#pragma unroll
                for (int r = 0; r < 8; r++) {
                    int m = mbase + r;
                    float v = alpha * (acc[i][j][r] + (bias ? bias[m] : 0.f));
                    pack[r] = (__bf16)v;
                }
                *(v8bf*)&Cb[(size_t)bz * sC + (size_t)nn * M + mbase] = pack;
            } else {
#pragma unroll
                for (int r = 0; r < 8; r++) {
                    int m = mbase + r;
                    float v = alpha * (acc[i][j][r] + (bias ? bias[m] : 0.f));
                    Cb[(size_t)bz * sC + (size_t)m * N + nn] = (__bf16)v;
                }
            }
        }
    }
}

// ------------------------------ K5: softmax --------------------------------
__global__ __launch_bounds__(256)
void softmax_kernel(__bf16* __restrict__ s) {
    __shared__ float red[256];
    const int t = threadIdx.x;
    const size_t base = (size_t)blockIdx.x * NDIM;

    v8bf pack = *(const v8bf*)&s[base + t * 8];
    float vals[8];
    float m = -3.4e38f;
#pragma unroll
    for (int i = 0; i < 8; i++) { vals[i] = (float)pack[i]; m = fmaxf(m, vals[i]); }
    red[t] = m;
    __syncthreads();
    for (int off = 128; off > 0; off >>= 1) {
        if (t < off) red[t] = fmaxf(red[t], red[t + off]);
        __syncthreads();
    }
    const float rowmax = red[0];
    __syncthreads();

    float sum = 0.f;
#pragma unroll
    for (int i = 0; i < 8; i++) { vals[i] = __expf(vals[i] - rowmax); sum += vals[i]; }
    red[t] = sum;
    __syncthreads();
    for (int off = 128; off > 0; off >>= 1) {
        if (t < off) red[t] += red[t + off];
        __syncthreads();
    }
    const float inv = 1.f / red[0];
#pragma unroll
    for (int i = 0; i < 8; i++) pack[i] = (__bf16)(vals[i] * inv);
    *(v8bf*)&s[base + t * 8] = pack;
}

// ------------------------------ launch -------------------------------------
extern "C" void kernel_launch(void* const* d_in, const int* in_sizes, int n_in,
                              void* d_out, int out_size, void* d_ws, size_t ws_size,
                              hipStream_t stream) {
    const float* x     = (const float*)d_in[0];
    const float* gamma = (const float*)d_in[1];
    const float* beta  = (const float*)d_in[2];
    const float* wq    = (const float*)d_in[3];
    const float* bq    = (const float*)d_in[4];
    const float* wk    = (const float*)d_in[5];
    const float* bk    = (const float*)d_in[6];
    const float* wv    = (const float*)d_in[7];
    const float* bv    = (const float*)d_in[8];
    const float* wo    = (const float*)d_in[9];
    const float* bo    = (const float*)d_in[10];
    float* out = (float*)d_out;

    const size_t WSZ = (size_t)CDIM * CDIM;
    const size_t HSZ = (size_t)BATCH * NDIM * CDIM;
    __bf16* w     = (__bf16*)d_ws;
    __bf16* wq_b  = w;
    __bf16* wk_b  = w + WSZ;
    __bf16* wv_b  = w + 2 * WSZ;
    __bf16* wo_b  = w + 3 * WSZ;
    __bf16* h_t   = w + 4 * WSZ;
    __bf16* q_t   = h_t + HSZ;
    __bf16* k_t   = q_t + HSZ;
    __bf16* vbuf  = k_t + HSZ;
    __bf16* o_t   = vbuf + HSZ;
    __bf16* s     = o_t + HSZ;

    const size_t sBN = (size_t)NDIM * CDIM;
    const size_t sSS = (size_t)NDIM * NDIM;
    const float qscale = 0.04419417382415922f;   // 512^-0.5

    wconv_kernel<<<(4 * (int)WSZ + 255) / 256, 256, 0, stream>>>(wq, wk, wv, wo, w, (int)WSZ);
    groupnorm_kernel<<<dim3(GROUPS, BATCH), 256, 0, stream>>>(x, gamma, beta, h_t);

    dim3 gProj(NDIM / BN, CDIM / BM, BATCH);    // (8, 4, 16)
    gemm_bf16_kernel<<<gProj, 256, 0, stream>>>(wq_b, 0, h_t, sBN, CDIM, NDIM, CDIM,
                                                bq, qscale, q_t, nullptr, nullptr, sBN, 0, 1);
    gemm_bf16_kernel<<<gProj, 256, 0, stream>>>(wk_b, 0, h_t, sBN, CDIM, NDIM, CDIM,
                                                bk, 1.f, k_t, nullptr, nullptr, sBN, 0, 1);
    gemm_bf16_kernel<<<gProj, 256, 0, stream>>>(wv_b, 0, h_t, sBN, CDIM, NDIM, CDIM,
                                                bv, 1.f, vbuf, nullptr, nullptr, sBN, 0, 0);

    dim3 gScore(NDIM / BN, NDIM / BM, BATCH);   // (8, 16, 16)
    gemm_bf16_kernel<<<gScore, 256, 0, stream>>>(q_t, sBN, k_t, sBN, NDIM, NDIM, CDIM,
                                                 nullptr, 1.f, s, nullptr, nullptr, sSS, 0, 0);

    softmax_kernel<<<BATCH * NDIM, 256, 0, stream>>>(s);

    gemm_bf16_kernel<<<gProj, 256, 0, stream>>>(vbuf, sBN, s, sSS, CDIM, NDIM, NDIM,
                                                nullptr, 1.f, o_t, nullptr, nullptr, sBN, 0, 1);

    gemm_bf16_kernel<<<gProj, 256, 0, stream>>>(wo_b, 0, o_t, sBN, CDIM, NDIM, CDIM,
                                                bo, 1.f, nullptr, out, x, sBN, sBN, 0);
}
